// MultiRelationGNN_16466904613366
// MI455X (gfx1250) — matrix operand order repaired and checked by reading, not compiled, not run on hardware
//
#include <hip/hip_runtime.h>

#define Nn 100000
#define Ee 1600000
#define Hh 32
#define Rr 8
#define PERM_N (Ee + 16 * Rr)

typedef __attribute__((ext_vector_type(16))) __bf16 v16bf;
typedef __attribute__((ext_vector_type(8)))  __bf16 v8bf;
typedef __attribute__((ext_vector_type(8)))  float  v8f;

__device__ __forceinline__ v8f wmma_bf(v16bf a, v16bf b, v8f c) {
  // D(f32 16x16) = A(bf16 16x32) * B(bf16 32x16) + C
  return __builtin_amdgcn_wmma_f32_16x16x32_bf16(false, a, false, b, (short)0, c,
                                                 false, false);
}

__device__ __forceinline__ v16bf zero16() {
  v16bf z;
#pragma unroll
  for (int i = 0; i < 16; ++i) z[i] = (__bf16)0.0f;
  return z;
}

__device__ __forceinline__ v8f zero8() {
  v8f z;
#pragma unroll
  for (int i = 0; i < 8; ++i) z[i] = 0.0f;
  return z;
}

// A fragment (16-bit A 16x32 layout): lane holds row (lane&15); half-wave hs=lane>>4:
// elems 0..7 = K hs*8..hs*8+7, elems 8..15 = K 16+hs*8.. . base = row_ptr + hs*8.
__device__ __forceinline__ v16bf load_a_bf(const __bf16* base) {
  v8bf lo = *(const v8bf*)(base);
  v8bf hi = *(const v8bf*)(base + 16);
  v16bf a;
#pragma unroll
  for (int i = 0; i < 8; ++i) { a[i] = lo[i]; a[i + 8] = hi[i]; }
  return a;
}

__device__ __forceinline__ v16bf load_a_f32(const float* base) {
  v16bf a;
#pragma unroll
  for (int i = 0; i < 8; ++i) {
    a[i]     = (__bf16)base[i];
    a[i + 8] = (__bf16)base[16 + i];
  }
  return a;
}

// B fragment, pre-laid-out: 512 bf16 per 32x16 tile, lane reads its 16 contiguous.
__device__ __forceinline__ v16bf load_b(const __bf16* tile, int lane) {
  const __bf16* p = tile + lane * 16;
  v8bf lo = *(const v8bf*)(p);
  v8bf hi = *(const v8bf*)(p + 8);
  v16bf b;
#pragma unroll
  for (int i = 0; i < 8; ++i) { b[i] = lo[i]; b[i + 8] = hi[i]; }
  return b;
}

// ---------------- setup kernels ----------------

__global__ void k_init(float* __restrict__ emb1, float* __restrict__ emb2,
                       int* __restrict__ perm, int* __restrict__ counts) {
  int i = blockIdx.x * blockDim.x + threadIdx.x;
  if (i < Nn * Hh) { emb1[i] = 0.0f; emb2[i] = 0.0f; }
  if (i < PERM_N) perm[i] = -1;
  if (i < Rr) counts[i] = 0;
}

__global__ void k_count(const int* __restrict__ et, int* __restrict__ counts) {
  int e = blockIdx.x * blockDim.x + threadIdx.x;
  if (e < Ee) atomicAdd(&counts[et[e]], 1);
}

__global__ void k_offsets(const int* __restrict__ counts, int* __restrict__ starts,
                          int* __restrict__ cursors) {
  if (threadIdx.x == 0 && blockIdx.x == 0) {
    int acc = 0;
    for (int r = 0; r < Rr; ++r) {
      starts[r] = acc;
      cursors[r] = acc;
      acc += (counts[r] + 15) & ~15;  // pad each relation bucket to 16
    }
    starts[Rr] = acc;
  }
}

__global__ void k_scatter(const int* __restrict__ et, int* __restrict__ cursors,
                          int* __restrict__ perm) {
  int e = blockIdx.x * blockDim.x + threadIdx.x;
  if (e < Ee) {
    int p = atomicAdd(&cursors[et[e]], 1);
    perm[p] = e;
  }
}

// Re-lay a row-major fp32 weight [mats][Kdim][Nfull] into bf16 WMMA B fragments.
// Tile = (mat, kh, nh); within tile: lane holds col nh*16+(lane&15),
// K = kh*32 + (e>>3)*16 + (lane>>4)*8 + (e&7)   (16 contiguous bf16 per lane).
__global__ __launch_bounds__(512) void k_wlay(const float* __restrict__ W,
                                              __bf16* __restrict__ out,
                                              int Kdim, int Nfull) {
  int nhc = Nfull >> 4;
  int tilesPerMat = (Kdim >> 5) * nhc;
  int mat = blockIdx.x / tilesPerMat;
  int tloc = blockIdx.x % tilesPerMat;
  int kh = tloc / nhc, nh = tloc % nhc;
  int lane = threadIdx.x >> 4, e = threadIdx.x & 15;
  int k = kh * 32 + (e >> 3) * 16 + (lane >> 4) * 8 + (e & 7);
  int n = nh * 16 + (lane & 15);
  out[(size_t)blockIdx.x * 512 + lane * 16 + e] =
      (__bf16)W[(size_t)(mat * Kdim + k) * Nfull + n];
}

__global__ void k_tobf(const float* __restrict__ in, __bf16* __restrict__ out) {
  int i = blockIdx.x * blockDim.x + threadIdx.x;
  if (i < Nn * Hh) out[i] = (__bf16)in[i];
}

// ---------------- WMMA compute kernels ----------------

// emb0 = x @ Wf + bf  (per wave: 16 nodes x 32 outputs, K loop over 128)
__global__ __launch_bounds__(256) void k_embed(const float* __restrict__ x,
                                               const __bf16* __restrict__ WfLay,
                                               const float* __restrict__ bfv,
                                               float* __restrict__ emb0,
                                               __bf16* __restrict__ emb0bf) {
  const int lane = threadIdx.x & 31;
  const int wid = threadIdx.x >> 5;
  const int tile = blockIdx.x * 8 + wid;
  const bool act = tile < (Nn / 16);
  const int node0 = (act ? tile : 0) * 16;
  const int lm = lane & 15;
  const int hs = lane >> 4;
  v8f c0 = zero8(), c1 = zero8();
#pragma unroll
  for (int kc = 0; kc < 4; ++kc) {
    const float* px = x + (size_t)(node0 + lm) * 128 + kc * 32 + hs * 8;
    v16bf a = load_a_f32(px);
    v16bf b0 = load_b(WfLay + (size_t)(kc * 2 + 0) * 512, lane);
    v16bf b1 = load_b(WfLay + (size_t)(kc * 2 + 1) * 512, lane);
    c0 = wmma_bf(a, b0, c0);
    c1 = wmma_bf(a, b1, c1);
  }
  if (act) {
#pragma unroll
    for (int nh = 0; nh < 2; ++nh) {
      const int n = nh * 16 + lm;
      const float bn = bfv[n];
#pragma unroll
      for (int j = 0; j < 8; ++j) {
        const int row = node0 + j + hs * 8;  // C layout: M = j + hs*8, N = lm
        const float v = (nh ? c1[j] : c0[j]) + bn;
        emb0[(size_t)row * Hh + n] = v;
        emb0bf[(size_t)row * Hh + n] = (__bf16)v;
      }
    }
  }
}

// One relational layer: each wave takes a 16-edge single-relation tile,
// msg = [x_j||x_i] @ W[rel] + b[rel], atomically scatter-summed into targets.
__global__ __launch_bounds__(256) void k_layer(const __bf16* __restrict__ embIn,
                                               float* __restrict__ embOut,
                                               const int* __restrict__ ei,
                                               const int* __restrict__ perm,
                                               const int* __restrict__ starts,
                                               const __bf16* __restrict__ Wlay,
                                               const float* __restrict__ bias) {
  const int lane = threadIdx.x & 31;
  const int wid = threadIdx.x >> 5;
  const int tile = blockIdx.x * 8 + wid;
  const int numTiles = starts[Rr] >> 4;
  const bool act = tile < numTiles;
  const int base = (act ? tile : 0) << 4;
  int rel = 0;
#pragma unroll
  for (int r = 1; r < Rr; ++r) rel += (base >= starts[r]) ? 1 : 0;
  const int lm = lane & 15;
  const int hs = lane >> 4;
  const int eA = perm[base + lm];  // -1 for bucket padding
  int coln = 0;
  v16bf a0, a1;
  if (eA >= 0) {
    const int rown = ei[eA];
    coln = ei[Ee + eA];
    a0 = load_a_bf(embIn + (size_t)rown * Hh + hs * 8);  // x_j -> K 0..31
    a1 = load_a_bf(embIn + (size_t)coln * Hh + hs * 8);  // x_i -> K 32..63
  } else {
    a0 = zero16();
    a1 = zero16();
  }
#pragma unroll
  for (int nh = 0; nh < 2; ++nh) {
    v16bf b0 = load_b(Wlay + (size_t)(rel * 4 + 0 * 2 + nh) * 512, lane);
    v16bf b1 = load_b(Wlay + (size_t)(rel * 4 + 1 * 2 + nh) * 512, lane);
    v8f c = zero8();
    c = wmma_bf(a0, b0, c);
    c = wmma_bf(a1, b1, c);
    const float bn = bias[rel * Hh + nh * 16 + lm];
#pragma unroll
    for (int j = 0; j < 8; ++j) {
      const int m = j + hs * 8;              // edge row of this C element
      const int em = __shfl(eA, m, 32);      // lane m gathered edge m's ids
      const int cm = __shfl(coln, m, 32);
      if (act && em >= 0)
        atomicAdd(&embOut[(size_t)cm * Hh + nh * 16 + lm], c[j] + bn);
    }
  }
}

// out = leaky( leaky([emb2||emb0] @ Wm1 + bm1) @ Wm2 + bm2 )
__global__ __launch_bounds__(256) void k_final(const float* __restrict__ emb2,
                                               const __bf16* __restrict__ emb0bf,
                                               const __bf16* __restrict__ Wm1Lay,
                                               const float* __restrict__ bm1,
                                               const __bf16* __restrict__ Wm2Lay,
                                               const float* __restrict__ bm2,
                                               float* __restrict__ out) {
  __shared__ float hbuf[8][16][32];
  const int lane = threadIdx.x & 31;
  const int wid = threadIdx.x >> 5;
  const int tile = blockIdx.x * 8 + wid;
  const bool act = tile < (Nn / 16);
  const int node0 = (act ? tile : 0) * 16;
  const int lm = lane & 15;
  const int hs = lane >> 4;

  v16bf aP = load_a_f32(emb2 + (size_t)(node0 + lm) * Hh + hs * 8);   // K 0..31
  v16bf aQ = load_a_bf(emb0bf + (size_t)(node0 + lm) * Hh + hs * 8);  // K 32..63
#pragma unroll
  for (int nh = 0; nh < 2; ++nh) {
    v16bf b0 = load_b(Wm1Lay + (size_t)(0 * 2 + nh) * 512, lane);
    v16bf b1 = load_b(Wm1Lay + (size_t)(1 * 2 + nh) * 512, lane);
    v8f c = zero8();
    c = wmma_bf(aP, b0, c);
    c = wmma_bf(aQ, b1, c);
#pragma unroll
    for (int j = 0; j < 8; ++j) {
      float v = c[j] + bm1[nh * 16 + lm];
      v = v > 0.0f ? v : 0.01f * v;
      hbuf[wid][j + hs * 8][nh * 16 + lm] = v;  // C layout -> row-major in LDS
    }
  }
  __syncthreads();
  v16bf a2;
#pragma unroll
  for (int i = 0; i < 8; ++i) {  // re-gather as A fragment
    a2[i]     = (__bf16)hbuf[wid][lm][hs * 8 + i];
    a2[i + 8] = (__bf16)hbuf[wid][lm][16 + hs * 8 + i];
  }
#pragma unroll
  for (int nh = 0; nh < 4; ++nh) {
    v16bf b = load_b(Wm2Lay + (size_t)nh * 512, lane);
    v8f c = zero8();
    c = wmma_bf(a2, b, c);
    if (act) {
#pragma unroll
      for (int j = 0; j < 8; ++j) {
        float v = c[j] + bm2[nh * 16 + lm];
        v = v > 0.0f ? v : 0.01f * v;
        out[(size_t)(node0 + j + hs * 8) * 64 + nh * 16 + lm] = v;
      }
    }
  }
}

// ---------------- host launcher ----------------

extern "C" void kernel_launch(void* const* d_in, const int* in_sizes, int n_in,
                              void* d_out, int out_size, void* d_ws, size_t ws_size,
                              hipStream_t stream) {
  const float* x   = (const float*)d_in[0];
  const int*   ei  = (const int*)d_in[1];
  const int*   et  = (const int*)d_in[2];
  // d_in[3] edge_time, d_in[4] pos: unused by the reference math
  const float* Wf  = (const float*)d_in[5];
  const float* bfv = (const float*)d_in[6];
  const float* W1  = (const float*)d_in[7];
  const float* b1  = (const float*)d_in[8];
  const float* W2  = (const float*)d_in[9];
  const float* b2  = (const float*)d_in[10];
  const float* Wm1 = (const float*)d_in[11];
  const float* bm1 = (const float*)d_in[12];
  const float* Wm2 = (const float*)d_in[13];
  const float* bm2 = (const float*)d_in[14];
  float* out = (float*)d_out;

  char* ws = (char*)d_ws;
  size_t off = 0;
  auto alloc = [&](size_t bytes) -> void* {
    void* p = ws + off;
    off = (off + bytes + 255) & ~(size_t)255;
    return p;
  };
  float*  emb0    = (float*)alloc((size_t)Nn * Hh * 4);
  float*  emb1    = (float*)alloc((size_t)Nn * Hh * 4);
  float*  emb2    = (float*)alloc((size_t)Nn * Hh * 4);
  __bf16* emb0bf  = (__bf16*)alloc((size_t)Nn * Hh * 2);
  __bf16* emb1bf  = (__bf16*)alloc((size_t)Nn * Hh * 2);
  int*    perm    = (int*)alloc((size_t)PERM_N * 4);
  int*    counts  = (int*)alloc(64);
  int*    cursors = (int*)alloc(64);
  int*    starts  = (int*)alloc(64);
  __bf16* WfLay   = (__bf16*)alloc(8 * 512 * 2);
  __bf16* W1Lay   = (__bf16*)alloc(32 * 512 * 2);
  __bf16* W2Lay   = (__bf16*)alloc(32 * 512 * 2);
  __bf16* Wm1Lay  = (__bf16*)alloc(4 * 512 * 2);
  __bf16* Wm2Lay  = (__bf16*)alloc(4 * 512 * 2);

  const int edgeBlocks  = Ee / 256;                 // 6250
  const int layerBlocks = (PERM_N / 16 + 7) / 8;    // 12501 (worst-case tiles)
  const int nodeBlocks  = (Nn / 16 + 7) / 8;        // 782

  k_init<<<(Nn * Hh) / 256, 256, 0, stream>>>(emb1, emb2, perm, counts);
  k_count<<<edgeBlocks, 256, 0, stream>>>(et, counts);
  k_offsets<<<1, 1, 0, stream>>>(counts, starts, cursors);
  k_scatter<<<edgeBlocks, 256, 0, stream>>>(et, cursors, perm);
  k_wlay<<<8, 512, 0, stream>>>(Wf, WfLay, 128, 32);
  k_wlay<<<32, 512, 0, stream>>>(W1, W1Lay, 64, 32);
  k_wlay<<<32, 512, 0, stream>>>(W2, W2Lay, 64, 32);
  k_wlay<<<4, 512, 0, stream>>>(Wm1, Wm1Lay, 64, 32);
  k_wlay<<<4, 512, 0, stream>>>(Wm2, Wm2Lay, 32, 64);
  k_embed<<<nodeBlocks, 256, 0, stream>>>(x, WfLay, bfv, emb0, emb0bf);
  k_layer<<<layerBlocks, 256, 0, stream>>>(emb0bf, emb1, ei, perm, starts, W1Lay, b1);
  k_tobf<<<(Nn * Hh) / 256, 256, 0, stream>>>(emb1, emb1bf);
  k_layer<<<layerBlocks, 256, 0, stream>>>(emb1bf, emb2, ei, perm, starts, W2Lay, b2);
  k_final<<<nodeBlocks, 256, 0, stream>>>(emb2, emb0bf, Wm1Lay, bm1, Wm2Lay, bm2, out);

  (void)in_sizes; (void)n_in; (void)out_size; (void)ws_size;
}
